// DyGMamba_18940805775844
// MI455X (gfx1250) — matrix operand order repaired and checked by reading, not compiled, not run on hardware
//
#include <hip/hip_runtime.h>
#include <math.h>

// ---------------- CDNA5 WMMA types ----------------
typedef __attribute__((ext_vector_type(16))) __bf16 v16bf;
typedef __attribute__((ext_vector_type(8)))  float  v8f;
typedef __attribute__((ext_vector_type(2)))  unsigned u32x2;

__device__ __forceinline__ unsigned short f2bf(float f) {
  unsigned u = __builtin_bit_cast(unsigned, f);
  unsigned r = u + 0x7FFFu + ((u >> 16) & 1u);   // round-to-nearest-even
  return (unsigned short)(r >> 16);
}
__device__ __forceinline__ __bf16 bfbits(unsigned short s) {
  return __builtin_bit_cast(__bf16, s);
}
__device__ __forceinline__ unsigned pk2(float lo, float hi) {
  return (unsigned)f2bf(lo) | ((unsigned)f2bf(hi) << 16);
}
__device__ __forceinline__ void put4(v16bf& f, int e, u32x2 v) {
  f[e]     = bfbits((unsigned short)(v.x & 0xFFFFu));
  f[e + 1] = bfbits((unsigned short)(v.x >> 16));
  f[e + 2] = bfbits((unsigned short)(v.y & 0xFFFFu));
  f[e + 3] = bfbits((unsigned short)(v.y >> 16));
}

#define ACT_NONE     0
#define ACT_GELU     1
#define ACT_SIGMOID  2
#define ACT_SOFTPLUS 3

__device__ __forceinline__ float apply_act(float v, int act) {
  switch (act) {
    case ACT_GELU: {
      float t = tanhf(0.7978845608028654f * (v + 0.044715f * v * v * v));
      return 0.5f * v * (1.f + t);
    }
    case ACT_SIGMOID:  return 1.f / (1.f + expf(-v));
    case ACT_SOFTPLUS: return (v > 20.f) ? v : log1pf(expf(v));
    default:           return v;
  }
}
__device__ __forceinline__ float silu(float v) { return v / (1.f + expf(-v)); }

// ---------------- Generic WMMA GEMM: C = act(A(M,K) @ W(N,K)^T + bias) [+ resid] ----------------
// Tile 64x64, 256 threads = 8 wave32 (4 x 2); each wave computes a 16(M) x 32(N) strip
// with 2x v_wmma_f32_16x16x32_bf16 per K-step. Double-buffered LDS software pipeline:
// global loads for tile k+1 issue before the WMMAs of tile k; one barrier per K-step.
__global__ void __launch_bounds__(256)
k_wmma_gemm(const float* __restrict__ A, int lda,
            const float* __restrict__ Wm,
            const float* __restrict__ bias,
            float* __restrict__ C, int ldc, int coff,
            const float* __restrict__ resid,
            int M, int N, int K, int act)
{
  __shared__ unsigned short lA[2][64][36];   // rows padded to 72B (8B multiple)
  __shared__ unsigned short lB[2][64][36];
  const int tid   = threadIdx.x;
  const int wave  = tid >> 5;
  const int lane  = tid & 31;
  const int blockM = blockIdx.y << 6;
  const int blockN = blockIdx.x << 6;
  const int waveM  = (wave & 3) << 4;     // 0,16,32,48
  const int waveN  = (wave >> 2) << 5;    // 0,32
  const int kh     = lane >> 4;           // lane half (0/1)
  const int l15    = lane & 15;
  // this thread's staging slots: 2 x (row, 4 consecutive cols)
  const int r0  = tid >> 3;               // rows 0..31
  const int r1  = r0 + 32;                // rows 32..63
  const int c4  = (tid & 7) << 2;         // col base (floats), multiple of 4

  const bool fullMN = (blockM + 64 <= M) && (blockN + 64 <= N);

  v8f acc0 = {0.f, 0.f, 0.f, 0.f, 0.f, 0.f, 0.f, 0.f};
  v8f acc1 = {0.f, 0.f, 0.f, 0.f, 0.f, 0.f, 0.f, 0.f};

  // -------- tile staging helpers (registers <-> global / LDS) --------
  auto load_tile = [&](int kt, float (&va)[8], float (&vb)[8]) {
    int k0 = kt << 5;
    if (fullMN && (k0 + 32 <= K)) {
      float4 a0 = *(const float4*)(A  + (size_t)(blockM + r0) * lda + k0 + c4);
      float4 a1 = *(const float4*)(A  + (size_t)(blockM + r1) * lda + k0 + c4);
      float4 b0 = *(const float4*)(Wm + (size_t)(blockN + r0) * K   + k0 + c4);
      float4 b1 = *(const float4*)(Wm + (size_t)(blockN + r1) * K   + k0 + c4);
      va[0] = a0.x; va[1] = a0.y; va[2] = a0.z; va[3] = a0.w;
      va[4] = a1.x; va[5] = a1.y; va[6] = a1.z; va[7] = a1.w;
      vb[0] = b0.x; vb[1] = b0.y; vb[2] = b0.z; vb[3] = b0.w;
      vb[4] = b1.x; vb[5] = b1.y; vb[6] = b1.z; vb[7] = b1.w;
    } else {
#pragma unroll
      for (int it = 0; it < 2; ++it) {
        int r = (it == 0) ? r0 : r1;
        int gm = blockM + r, gn = blockN + r;
#pragma unroll
        for (int e = 0; e < 4; ++e) {
          int gk = k0 + c4 + e;
          va[it * 4 + e] = (gm < M && gk < K) ? A[(size_t)gm * lda + gk] : 0.f;
          vb[it * 4 + e] = (gn < N && gk < K) ? Wm[(size_t)gn * K + gk] : 0.f;
        }
      }
    }
  };
  auto store_tile = [&](int buf, const float (&va)[8], const float (&vb)[8]) {
    u32x2 w0; w0.x = pk2(va[0], va[1]); w0.y = pk2(va[2], va[3]);
    *(u32x2*)&lA[buf][r0][c4] = w0;
    u32x2 w1; w1.x = pk2(va[4], va[5]); w1.y = pk2(va[6], va[7]);
    *(u32x2*)&lA[buf][r1][c4] = w1;
    u32x2 w2; w2.x = pk2(vb[0], vb[1]); w2.y = pk2(vb[2], vb[3]);
    *(u32x2*)&lB[buf][r0][c4] = w2;
    u32x2 w3; w3.x = pk2(vb[4], vb[5]); w3.y = pk2(vb[6], vb[7]);
    *(u32x2*)&lB[buf][r1][c4] = w3;
  };

  // -------- software-pipelined main loop --------
  const int nk = (K + 31) >> 5;
  {
    float ra[8], rb[8];
    load_tile(0, ra, rb);
    store_tile(0, ra, rb);
  }
  __syncthreads();

  for (int kt = 0; kt < nk; ++kt) {
    const int cur = kt & 1;
    float na[8], nb[8];
    const bool more = (kt + 1 < nk);
    if (more) load_tile(kt + 1, na, nb);   // global loads issue; no wait yet

    // Fragments per ISA 7.12.2: A (16-bit 16x32) interleaved K-pairs; B in the same K order as A.
    v16bf fa, fb0, fb1;
    {
      const unsigned short* pa = &lA[cur][waveM + l15][0];
      put4(fa, 0,  *(const u32x2*)(pa + kh * 8));
      put4(fa, 4,  *(const u32x2*)(pa + kh * 8 + 4));
      put4(fa, 8,  *(const u32x2*)(pa + 16 + kh * 8));
      put4(fa, 12, *(const u32x2*)(pa + 16 + kh * 8 + 4));
      const unsigned short* pb0 = &lB[cur][waveN + l15][0];
      const unsigned short* pb1 = &lB[cur][waveN + 16 + l15][0];
#pragma unroll
      for (int q = 0; q < 4; ++q) {
        put4(fb0, q * 4, *(const u32x2*)(pb0 + (q >> 1) * 16 + kh * 8 + (q & 1) * 4));
        put4(fb1, q * 4, *(const u32x2*)(pb1 + (q >> 1) * 16 + kh * 8 + (q & 1) * 4));
      }
    }
    acc0 = __builtin_amdgcn_wmma_f32_16x16x32_bf16(false, fa, false, fb0,
                                                   (short)0, acc0, false, false);
    acc1 = __builtin_amdgcn_wmma_f32_16x16x32_bf16(false, fa, false, fb1,
                                                   (short)0, acc1, false, false);

    if (more) store_tile(cur ^ 1, na, nb); // waits for the prefetch loads here
    __syncthreads();
  }

  // Epilogue: C/D layout — VGPR r: M=r (lanes 0-15), M=r+8 (lanes 16-31); N=lane&15
  int gn0 = blockN + waveN + l15;
  int gn1 = gn0 + 16;
  float bv0 = (bias && gn0 < N) ? bias[gn0] : 0.f;
  float bv1 = (bias && gn1 < N) ? bias[gn1] : 0.f;
#pragma unroll
  for (int r = 0; r < 8; ++r) {
    int gm = blockM + waveM + r + (kh << 3);
    if (gm < M) {
      size_t rowoff = (size_t)gm * ldc + coff;
      if (gn0 < N) {
        float v = apply_act(acc0[r] + bv0, act);
        if (resid) v += resid[rowoff + gn0];
        C[rowoff + gn0] = v;
      }
      if (gn1 < N) {
        float v = apply_act(acc1[r] + bv1, act);
        if (resid) v += resid[rowoff + gn1];
        C[rowoff + gn1] = v;
      }
    }
  }
}

// ---------------- Elementwise / scan / reduction kernels ----------------
__global__ void k_time_enc(const float* __restrict__ t, const float* __restrict__ w,
                           const float* __restrict__ b, float* __restrict__ out,
                           int n, int F) {
  int i = blockIdx.x * blockDim.x + threadIdx.x;
  if (i >= n * F) return;
  int r = i / F, f = i - r * F;
  out[i] = cosf(t[r] * w[f] + b[f]);
}

__global__ void k_nif_h(const float* __restrict__ a, const float* __restrict__ w1,
                        const float* __restrict__ b1, float* __restrict__ h, int n) {
  int i = blockIdx.x * blockDim.x + threadIdx.x;
  if (i >= n) return;
  int c = i & 127;
  int rj = i >> 7;
  float v = a[rj] * w1[c] + b1[c];
  h[i] = v > 0.f ? v : 0.f;
}

__global__ void k_pairsum(const float* __restrict__ t, float* __restrict__ o,
                          int total, int C) {
  int i = blockIdx.x * blockDim.x + threadIdx.x;
  if (i >= total) return;
  int r = i / C, c = i - r * C;
  o[i] = t[(size_t)(2 * r) * C + c] + t[(size_t)(2 * r + 1) * C + c];
}

// depthwise causal conv (D_CONV=4) + SiLU
__global__ void k_conv_silu(const float* __restrict__ xi, int ldxi,
                            const float* __restrict__ w, const float* __restrict__ cb,
                            float* __restrict__ xc, int Bb, int Ls, int d) {
  int i = blockIdx.x * blockDim.x + threadIdx.x;
  if (i >= Bb * Ls * d) return;
  int dd = i % d;
  int bt = i / d;
  int t = bt % Ls, b = bt / Ls;
  float s = cb[dd];
#pragma unroll
  for (int k = 0; k < 4; ++k) {
    int tt = t - 3 + k;
    if (tt >= 0) s += xi[(size_t)(b * Ls + tt) * ldxi + dd] * w[dd * 4 + k];
  }
  xc[i] = silu(s);
}

// selective scan over L with 16-wide state, fused +D*xc and *silu(z) gating
__global__ void k_scan(const float* __restrict__ dt, const float* __restrict__ xc,
                       const float* __restrict__ xdbl, int lddbl, int dtr,
                       const float* __restrict__ A_log, const float* __restrict__ Dp,
                       const float* __restrict__ z, int ldz,
                       float* __restrict__ y, int ldy,
                       int Bb, int Ls, int d) {
  int i = blockIdx.x * blockDim.x + threadIdx.x;
  if (i >= Bb * d) return;
  int dd = i % d, b = i / d;
  float Arow[16], h[16];
#pragma unroll
  for (int s = 0; s < 16; ++s) { Arow[s] = -expf(A_log[dd * 16 + s]); h[s] = 0.f; }
  float Dv = Dp[dd];
  for (int t = 0; t < Ls; ++t) {
    size_t row = (size_t)b * Ls + t;
    float dtv = dt[row * d + dd];
    float xcv = xc[row * d + dd];
    const float* db = xdbl + row * lddbl;
    float accv = 0.f;
#pragma unroll
    for (int s = 0; s < 16; ++s) {
      float Bm = db[dtr + s];
      float Cm = db[dtr + 16 + s];
      h[s] = expf(dtv * Arow[s]) * h[s] + dtv * Bm * xcv;
      accv += h[s] * Cm;
    }
    float zv = z[row * ldz + dd];
    y[row * ldy + dd] = (accv + Dv * xcv) * silu(zv);
  }
}

__global__ void __launch_bounds__(256)
k_layernorm(const float* __restrict__ x, const float* __restrict__ g,
            const float* __restrict__ bta, float* __restrict__ o, int D) {
  __shared__ float red[256];
  size_t row = blockIdx.x;
  const float* xr = x + row * D;
  float s = 0.f;
  for (int i = threadIdx.x; i < D; i += 256) s += xr[i];
  red[threadIdx.x] = s; __syncthreads();
  for (int off = 128; off > 0; off >>= 1) {
    if (threadIdx.x < off) red[threadIdx.x] += red[threadIdx.x + off];
    __syncthreads();
  }
  float mu = red[0] / D; __syncthreads();
  float v = 0.f;
  for (int i = threadIdx.x; i < D; i += 256) { float dd = xr[i] - mu; v += dd * dd; }
  red[threadIdx.x] = v; __syncthreads();
  for (int off = 128; off > 0; off >>= 1) {
    if (threadIdx.x < off) red[threadIdx.x] += red[threadIdx.x + off];
    __syncthreads();
  }
  float inv = rsqrtf(red[0] / D + 1e-5f);
  for (int i = threadIdx.x; i < D; i += 256)
    o[row * D + i] = (xr[i] - mu) * inv * g[i] + bta[i];
}

// scores[b,l] = dot(x[b,l,:], sel[b,:]*wagg_w) + wagg_b  (one wave32 per row)
__global__ void __launch_bounds__(128)
k_scores(const float* __restrict__ x, const float* __restrict__ sel,
         const float* __restrict__ ww, const float* __restrict__ wb,
         float* __restrict__ scores, int Bb, int Ls, int D) {
  int wid = (blockIdx.x * blockDim.x + threadIdx.x) >> 5;
  int lane = threadIdx.x & 31;
  if (wid >= Bb * Ls) return;
  int b = wid / Ls;
  const float* xr = x + (size_t)wid * D;
  const float* sr = sel + (size_t)b * D;
  float s = 0.f;
  for (int dd = lane; dd < D; dd += 32) s += xr[dd] * sr[dd] * ww[dd];
  for (int off = 16; off > 0; off >>= 1) s += __shfl_down(s, off, 32);
  if (lane == 0) scores[wid] = s + wb[0];
}

__global__ void __launch_bounds__(256)
k_softmax(const float* __restrict__ s, float* __restrict__ a, int Ls) {
  __shared__ float red[256];
  int b = blockIdx.x;
  const float* sr = s + (size_t)b * Ls;
  float m = -3.4e38f;
  for (int i = threadIdx.x; i < Ls; i += 256) m = fmaxf(m, sr[i]);
  red[threadIdx.x] = m; __syncthreads();
  for (int off = 128; off > 0; off >>= 1) {
    if (threadIdx.x < off) red[threadIdx.x] = fmaxf(red[threadIdx.x], red[threadIdx.x + off]);
    __syncthreads();
  }
  m = red[0]; __syncthreads();
  float sum = 0.f;
  for (int i = threadIdx.x; i < Ls; i += 256) {
    float e = expf(sr[i] - m);
    a[(size_t)b * Ls + i] = e;
    sum += e;
  }
  red[threadIdx.x] = sum; __syncthreads();
  for (int off = 128; off > 0; off >>= 1) {
    if (threadIdx.x < off) red[threadIdx.x] += red[threadIdx.x + off];
    __syncthreads();
  }
  float inv = 1.f / red[0];
  for (int i = threadIdx.x; i < Ls; i += 256) a[(size_t)b * Ls + i] *= inv;
}

__global__ void k_pool(const float* __restrict__ attn, const float* __restrict__ x,
                       float* __restrict__ pooled, int Bb, int Ls, int D) {
  int i = blockIdx.x * blockDim.x + threadIdx.x;
  if (i >= Bb * D) return;
  int dd = i % D, b = i / D;
  float s = 0.f;
  for (int l = 0; l < Ls; ++l)
    s += attn[(size_t)b * Ls + l] * x[(size_t)(b * Ls + l) * D + dd];
  pooled[i] = s;
}

// ---------------- Host side ----------------
static inline int cdiv(long long a, int b) { return (int)((a + b - 1) / b); }

static void gemm(hipStream_t s, const float* A, int lda, const float* Wm,
                 const float* bias, float* C, int ldc, int coff,
                 const float* resid, int M, int N, int K, int act) {
  dim3 grid((N + 63) / 64, (M + 63) / 64);
  k_wmma_gemm<<<grid, dim3(256), 0, s>>>(A, lda, Wm, bias, C, ldc, coff, resid,
                                         M, N, K, act);
}

struct MambaP {
  const float *in_proj, *conv_w, *conv_b, *x_proj, *dt_proj, *dt_bias, *A_log, *D, *out_proj;
};
static MambaP mk_mamba(void* const* d_in, int base) {
  MambaP m;
  m.in_proj  = (const float*)d_in[base + 0];
  m.conv_w   = (const float*)d_in[base + 1];
  m.conv_b   = (const float*)d_in[base + 2];
  m.x_proj   = (const float*)d_in[base + 3];
  m.dt_proj  = (const float*)d_in[base + 4];
  m.dt_bias  = (const float*)d_in[base + 5];
  m.A_log    = (const float*)d_in[base + 6];
  m.D        = (const float*)d_in[base + 7];
  m.out_proj = (const float*)d_in[base + 8];
  return m;
}

static void run_mamba(hipStream_t s, const float* xin, float* xout,
                      float* xz, float* xc, float* xdbl, float* dtb,
                      int Bb, int Ls, int d, int dtr, const MambaP& p) {
  int Mrows = Bb * Ls;
  int d2 = 2 * d;
  int nd = dtr + 32;
  // in_proj (no bias): xz = x @ in_proj^T
  gemm(s, xin, d, p.in_proj, nullptr, xz, d2, 0, nullptr, Mrows, d2, d, ACT_NONE);
  // depthwise conv + silu on xi = xz[:, :d]
  k_conv_silu<<<cdiv((long long)Mrows * d, 256), 256, 0, s>>>(xz, d2, p.conv_w, p.conv_b,
                                                              xc, Bb, Ls, d);
  // x_proj (no bias)
  gemm(s, xc, d, p.x_proj, nullptr, xdbl, nd, 0, nullptr, Mrows, nd, d, ACT_NONE);
  // dt = softplus(dtr-part @ dt_proj^T + dt_bias)
  gemm(s, xdbl, nd, p.dt_proj, p.dt_bias, dtb, d, 0, nullptr, Mrows, d, dtr, ACT_SOFTPLUS);
  // scan (+D*xc, *silu(z)); write gated y back into xz[:, :d] (xi dead)
  k_scan<<<cdiv((long long)Bb * d, 256), 256, 0, s>>>(dtb, xc, xdbl, nd, dtr, p.A_log,
                                                      p.D, xz + d, d2, xz, d2, Bb, Ls, d);
  // out_proj (no bias)
  gemm(s, xz, d2, p.out_proj, nullptr, xout, d, 0, nullptr, Mrows, d, d, ACT_NONE);
}

extern "C" void kernel_launch(void* const* d_in, const int* in_sizes, int n_in,
                              void* d_out, int out_size, void* d_ws, size_t ws_size,
                              hipStream_t stream) {
  const int Bh = 128, Lh = 512, Th = 10, CEDh = 128, DMh = 512, GDh = 64;
  const int NFh = 172, TFh = 100;
  const size_t BL = (size_t)Bh * Lh;   // 65536
  const size_t BT = (size_t)Bh * Th;   // 1280

  const float* node_feats = (const float*)d_in[0];
  const float* edge_feats = (const float*)d_in[1];
  const float* times      = (const float*)d_in[2];
  const float* appear     = (const float*)d_in[3];
  const float* t_diff     = (const float*)d_in[4];
  // params flattened in setup_inputs insertion order
  const float* time_w  = (const float*)d_in[5];
  const float* time_b  = (const float*)d_in[6];
  const float* nif1_w  = (const float*)d_in[7];
  const float* nif1_b  = (const float*)d_in[8];
  const float* nif2_w  = (const float*)d_in[9];
  const float* nif2_b  = (const float*)d_in[10];
  const float* pn_w = (const float*)d_in[11]; const float* pn_b = (const float*)d_in[12];
  const float* pe_w = (const float*)d_in[13]; const float* pe_b = (const float*)d_in[14];
  const float* pt_w = (const float*)d_in[15]; const float* pt_b = (const float*)d_in[16];
  const float* pf_w = (const float*)d_in[17]; const float* pf_b = (const float*)d_in[18];
  const float* red_w = (const float*)d_in[19]; const float* red_b = (const float*)d_in[20];
  MambaP m0  = mk_mamba(d_in, 21);
  MambaP m1  = mk_mamba(d_in, 30);
  MambaP mt0 = mk_mamba(d_in, 39);
  MambaP mt1 = mk_mamba(d_in, 48);
  const float* ln_g = (const float*)d_in[57];
  const float* ln_b = (const float*)d_in[58];
  const float* ffn1_w = (const float*)d_in[59]; const float* ffn1_b = (const float*)d_in[60];
  const float* ffn2_w = (const float*)d_in[61]; const float* ffn2_b = (const float*)d_in[62];
  const float* ptd_w = (const float*)d_in[63]; const float* ptd_b = (const float*)d_in[64];
  const float* ptdup_w = (const float*)d_in[65]; const float* ptdup_b = (const float*)d_in[66];
  const float* sel_w = (const float*)d_in[67]; const float* sel_b = (const float*)d_in[68];
  const float* wagg_w = (const float*)d_in[69]; const float* wagg_b = (const float*)d_in[70];
  const float* out_w = (const float*)d_in[71]; const float* out_b = (const float*)d_in[72];
  const float* outtd_w = (const float*)d_in[73]; const float* outtd_b = (const float*)d_in[74];

  // ---- workspace layout (floats); phase-scoped arena reuse ----
  float* w = (float*)d_ws;
  float* xA = w;                         // BL*DM
  float* xB = xA + BL * DMh;             // BL*DM
  float* arena = xB + BL * DMh;

  // Phase 1: features
  float* time_f = arena;                 // BL*TF
  float* hbuf   = time_f + BL * TFh;     // BL*2*CED
  float* niftmp = hbuf + BL * 2 * CEDh;  // BL*2*CED
  float* nifb   = niftmp + BL * 2 * CEDh;// BL*CED
  // Phase 2: mamba (d=512)
  float* xz   = arena;                   // BL*1024
  float* xcb  = xz + BL * 2 * DMh;       // BL*512
  float* xdbl = xcb + BL * DMh;          // BL*64
  float* dtb  = xdbl + BL * 64;          // BL*512
  // Phase 3: LN + FFN
  float* xn  = arena;                    // BL*512
  float* hid = xn + BL * DMh;            // 8192*2048
  // Phase 4/5: td path + attention
  float* tdf   = arena;                  // BT*TF
  float* td1   = tdf + BT * TFh;         // BT*GD
  float* td2   = td1 + BT * GDh;         // BT*GD
  float* xz2   = td2 + BT * GDh;         // BT*2*GD
  float* xc2   = xz2 + BT * 2 * GDh;     // BT*GD
  float* xdbl2 = xc2 + BT * GDh;         // BT*36
  float* dt2   = xdbl2 + BT * 36;        // BT*GD
  float* tdup  = dt2 + BT * GDh;         // B*DM
  float* selb  = tdup + (size_t)Bh * DMh;// B*DM
  float* scores = selb + (size_t)Bh * DMh; // B*L
  float* attn   = scores + BL;           // B*L
  float* pooled = attn + BL;             // B*DM

  // ======== Phase 1: time enc, nif, projections, reduce ========
  k_time_enc<<<cdiv(BL * TFh, 256), 256, 0, stream>>>(times, time_w, time_b, time_f,
                                                      (int)BL, TFh);
  k_nif_h<<<cdiv(BL * 2 * CEDh, 256), 256, 0, stream>>>(appear, nif1_w, nif1_b, hbuf,
                                                        (int)(BL * 2 * CEDh));
  gemm(stream, hbuf, CEDh, nif2_w, nif2_b, niftmp, CEDh, 0, nullptr,
       (int)(BL * 2), CEDh, CEDh, ACT_NONE);
  k_pairsum<<<cdiv(BL * CEDh, 256), 256, 0, stream>>>(niftmp, nifb, (int)(BL * CEDh), CEDh);

  gemm(stream, node_feats, NFh, pn_w, pn_b, xA, DMh, 0,   nullptr, (int)BL, CEDh, NFh, ACT_NONE);
  gemm(stream, edge_feats, NFh, pe_w, pe_b, xA, DMh, 128, nullptr, (int)BL, CEDh, NFh, ACT_NONE);
  gemm(stream, time_f, TFh, pt_w, pt_b, xA, DMh, 256, nullptr, (int)BL, CEDh, TFh, ACT_NONE);
  gemm(stream, nifb, CEDh, pf_w, pf_b, xA, DMh, 384, nullptr, (int)BL, CEDh, CEDh, ACT_NONE);
  gemm(stream, xA, DMh, red_w, red_b, xB, DMh, 0, nullptr, (int)BL, DMh, DMh, ACT_NONE);

  // ======== Phase 2: 2x Mamba (d=512, dt_rank=32) ========
  run_mamba(stream, xB, xA, xz, xcb, xdbl, dtb, Bh, Lh, DMh, 32, m0);
  run_mamba(stream, xA, xB, xz, xcb, xdbl, dtb, Bh, Lh, DMh, 32, m1);
  float* x = xB;

  // ======== Phase 3: LN + FFN (M-chunked, residual fused) ========
  k_layernorm<<<(int)BL, 256, 0, stream>>>(x, ln_g, ln_b, xn, DMh);
  for (int mo = 0; mo < (int)BL; mo += 8192) {
    gemm(stream, xn + (size_t)mo * DMh, DMh, ffn1_w, ffn1_b, hid, 4 * DMh, 0, nullptr,
         8192, 4 * DMh, DMh, ACT_GELU);
    gemm(stream, hid, 4 * DMh, ffn2_w, ffn2_b, x + (size_t)mo * DMh, DMh, 0,
         x + (size_t)mo * DMh, 8192, DMh, 4 * DMh, ACT_NONE);
  }

  // ======== Phase 4: td path (d=64, dt_rank=4) ========
  k_time_enc<<<cdiv(BT * TFh, 256), 256, 0, stream>>>(t_diff, time_w, time_b, tdf,
                                                      (int)BT, TFh);
  gemm(stream, tdf, TFh, ptd_w, ptd_b, td1, GDh, 0, nullptr, (int)BT, GDh, TFh, ACT_NONE);
  run_mamba(stream, td1, td2, xz2, xc2, xdbl2, dt2, Bh, Th, GDh, 4, mt0);
  run_mamba(stream, td2, td1, xz2, xc2, xdbl2, dt2, Bh, Th, GDh, 4, mt1);
  const float* td_last = td1 + (size_t)(Th - 1) * GDh;  // row b at stride T*GD
  const int ld_tdlast = Th * GDh;                       // 640

  gemm(stream, td_last, ld_tdlast, ptdup_w, ptdup_b, tdup, DMh, 0, nullptr,
       Bh, DMh, GDh, ACT_NONE);
  gemm(stream, tdup, DMh, sel_w, sel_b, selb, DMh, 0, nullptr,
       Bh, DMh, DMh, ACT_SIGMOID);

  // ======== Phase 5: selective attention pooling + output head ========
  k_scores<<<cdiv((long long)Bh * Lh, 4), 128, 0, stream>>>(x, selb, wagg_w, wagg_b,
                                                            scores, Bh, Lh, DMh);
  k_softmax<<<Bh, 256, 0, stream>>>(scores, attn, Lh);
  k_pool<<<cdiv((long long)Bh * DMh, 256), 256, 0, stream>>>(attn, x, pooled, Bh, Lh, DMh);

  float* outp = (float*)d_out;
  gemm(stream, pooled, DMh, out_w, out_b, outp, NFh, 0, nullptr, Bh, NFh, DMh, ACT_NONE);
  gemm(stream, td_last, ld_tdlast, outtd_w, outtd_b, outp, NFh, 0, outp,
       Bh, NFh, GDh, ACT_NONE);
}
